// CausalSelfAttention_50603304681961
// MI455X (gfx1250) — matrix operand order repaired and checked by reading, not compiled, not run on hardware
//
#include <hip/hip_runtime.h>

// ---------------------------------------------------------------------------
// CDNA5 (gfx1250) causal self-attention, bf16 WMMA + TDM path.
//   B=2 T=2048 C=1024 H=16 D=64.
// Pipeline (all on `stream`):
//   1) cvt x -> bf16 row-major [B*T, C]
//   2) cvt W_qkv -> bf16 transposed [3C, C]   (k contiguous per output row)
//   3) cvt W_proj -> bf16 transposed [C, C]
//   4) QKV GEMM: TDM double-buffered LDS staging, 8 waves/block, 128x256 macro
//      tile, 64x64 wmma tile per wave -> q[B,H,T,D], k[B,H,T,D], vT[B,H,D,T]
//   5) transposed flash attention (S^T = K*Q^T, O^T = V^T*P^T) -> y bf16 [B,T,C]
//   6) proj GEMM (same TDM scheme) + bias -> fp32 out
// ---------------------------------------------------------------------------

#define DEV __device__ __forceinline__

typedef __attribute__((ext_vector_type(16))) __bf16 bf16x16;
typedef __attribute__((ext_vector_type(8)))  __bf16 bf16x8;
typedef __attribute__((ext_vector_type(8)))  float  f32x8;
typedef __attribute__((ext_vector_type(4)))  unsigned int u32x4;
typedef __attribute__((ext_vector_type(4)))  int i32x4;
typedef __attribute__((ext_vector_type(8)))  int i32x8;

static constexpr int Bk = 2, Tk = 2048, Ck = 1024, Hk = 16, Dk = 64;

DEV unsigned short f2bf(float f) {
  unsigned u = __builtin_bit_cast(unsigned, f);
  unsigned r = u + 0x7FFFu + ((u >> 16) & 1u);      // round-to-nearest-even
  if ((u & 0x7F800000u) == 0x7F800000u) r = u;      // inf/nan passthrough
  return (unsigned short)(r >> 16);
}
DEV __bf16 f2bfh(float f) { return __builtin_bit_cast(__bf16, f2bf(f)); }

DEV bf16x16 join8(bf16x8 lo, bf16x8 hi) {
  return __builtin_shufflevector(lo, hi, 0,1,2,3,4,5,6,7,8,9,10,11,12,13,14,15);
}

// A-fragment (16x32, 16-bit): lane L holds row = row0 + L%16,
// elements i -> col = col0 + (i>>3)*16 + 8*(L>>4) + (i&7)  (two contiguous 16B chunks)
DEV bf16x16 load_a_frag(const unsigned short* p, int ld, int row0, int col0, int lane) {
  const unsigned short* r = p + (row0 + (lane & 15)) * ld + col0 + 8 * (lane >> 4);
  bf16x8 lo = *(const bf16x8*)(r);
  bf16x8 hi = *(const bf16x8*)(r + 16);
  return join8(lo, hi);
}

// B-fragment (32x16, 16-bit): lane L holds col n = n0 + L%16,
// elements i -> k = k0 + (i>>3)*16 + 8*(L>>4) + (i&7)  (two contiguous 16B chunks, src layout p[n*ld + k])
DEV bf16x16 load_b_frag(const unsigned short* p, int ld, int n0, int k0, int lane) {
  const unsigned short* r = p + (n0 + (lane & 15)) * ld + k0 + 8 * (lane >> 4);
  bf16x8 lo = *(const bf16x8*)(r);
  bf16x8 hi = *(const bf16x8*)(r + 16);
  return join8(lo, hi);
}

// LDS slab variants: slab is rows x 32 bf16, row-major.
DEV bf16x16 lds_a_frag(const unsigned short* slab, int row0, int lane) {
  const unsigned short* r = slab + (row0 + (lane & 15)) * 32 + 8 * (lane >> 4);
  bf16x8 lo = *(const bf16x8*)(r);
  bf16x8 hi = *(const bf16x8*)(r + 16);
  return join8(lo, hi);
}
DEV bf16x16 lds_b_frag(const unsigned short* slab, int row0, int lane) {
  const unsigned short* r = slab + (row0 + (lane & 15)) * 32 + 8 * (lane >> 4);
  bf16x8 lo = *(const bf16x8*)(r);
  bf16x8 hi = *(const bf16x8*)(r + 16);
  return join8(lo, hi);
}

DEV f32x8 wmma_bf16(bf16x16 a, bf16x16 b, f32x8 c) {
  return __builtin_amdgcn_wmma_f32_16x16x32_bf16(false, a, false, b, (short)0, c, false, false);
}

// ---------------------------------------------------------------------------
// Tensor Data Mover: 2D tile (tile_k x tile_rows, bf16) global -> LDS.
// D# packed per CDNA5 ISA; groups 2/3 zero (2D).
// Generic flat LDS address truncated to 32 bits == LDS byte offset (aperture map).
// ---------------------------------------------------------------------------
DEV void tdm_load_2d(const void* lds_dst, const void* gsrc,
                     unsigned tile_k, unsigned tile_rows, unsigned row_stride_elems) {
  unsigned long long ga = (unsigned long long)(size_t)gsrc;
  unsigned lds_addr = (unsigned)(unsigned long long)(size_t)lds_dst;
  u32x4 g0;
  g0[0] = 1u;                                            // count=1, user descriptor
  g0[1] = lds_addr;                                      // lds byte address
  g0[2] = (unsigned)(ga & 0xFFFFFFFFu);                  // global_addr[31:0]
  g0[3] = (unsigned)((ga >> 32) & 0x01FFFFFFu) | 0x80000000u;  // addr[56:32] | type=2
  unsigned td0 = row_stride_elems;                       // tensor dim0 (elements)
  unsigned td1 = 1u << 20;                               // tensor dim1 (large, interior tile)
  unsigned long long st0 = row_stride_elems;             // dim0 stride (elements)
  i32x8 g1;
  g1[0] = (int)(1u << 16);                               // data_size = 1 -> 2 bytes
  g1[1] = (int)((td0 & 0xFFFFu) << 16);                  // tensor_dim0[15:0]
  g1[2] = (int)((td0 >> 16) | ((td1 & 0xFFFFu) << 16));  // dim0[31:16] | dim1[15:0]
  g1[3] = (int)((td1 >> 16) | (tile_k << 16));           // dim1[31:16] | tile_dim0
  g1[4] = (int)(tile_rows & 0xFFFFu);                    // tile_dim1 (tile_dim2 = 0)
  g1[5] = (int)(st0 & 0xFFFFFFFFull);                    // dim0_stride[31:0]
  g1[6] = (int)(st0 >> 32);                              // dim0_stride[47:32]
  g1[7] = 0;
  i32x4 z4 = {};
#if defined(__clang_major__) && (__clang_major__ >= 23)
  i32x8 z8 = {};
  __builtin_amdgcn_tensor_load_to_lds(g0, g1, z4, z4, z8, 0);
#else
  __builtin_amdgcn_tensor_load_to_lds(g0, g1, z4, z4, 0);
#endif
}

// ---------------------------------------------------------------------------
// Conversion kernels
// ---------------------------------------------------------------------------
__global__ void cvt_bf16_kernel(const float* __restrict__ x,
                                unsigned short* __restrict__ xb, int n) {
  int i = blockIdx.x * blockDim.x + threadIdx.x;
  if (i < n) xb[i] = f2bf(x[i]);
}

// w[K][N] (row-major) -> wT[N][K] bf16. Thread i -> (n = i/K, k = i%K), write coalesced.
__global__ void cvt_transpose_kernel(const float* __restrict__ w,
                                     unsigned short* __restrict__ wT, int K, int N) {
  int i = blockIdx.x * blockDim.x + threadIdx.x;
  if (i >= K * N) return;
  int k = i % K, n = i / K;
  wT[i] = f2bf(w[k * N + n]);
}

// ---------------------------------------------------------------------------
// QKV GEMM: [4096,1024]x[1024,3072] + bias, TDM-staged operands,
// scatter into q/k ([B,H,T,D]) and vT ([B,H,D,T]).
// Block: 8 waves (2M x 4N), macro tile 128x256, wave tile 64x64.
// ---------------------------------------------------------------------------
__global__ __launch_bounds__(256) void qkv_gemm_kernel(
    const unsigned short* __restrict__ xb, const unsigned short* __restrict__ wT,
    const float* __restrict__ bias,
    unsigned short* __restrict__ qb, unsigned short* __restrict__ kb,
    unsigned short* __restrict__ vT) {
  __shared__ unsigned short ldsA[2][128 * 32];    // 16 KB
  __shared__ unsigned short ldsB[2][256 * 32];    // 32 KB
  const int lane = threadIdx.x & 31;
  const int wv   = threadIdx.x >> 5;
  const int NTB = (3 * Ck) / 256;                 // 12
  const int bm = blockIdx.x / NTB, bn = blockIdx.x % NTB;
  const int m0 = bm * 128, n0 = bn * 256;
  const int wm = (wv >> 2) * 64, wn = (wv & 3) * 64;

  if (wv == 0) {
    tdm_load_2d(ldsA[0], xb + (size_t)m0 * Ck, 32, 128, Ck);
    tdm_load_2d(ldsB[0], wT + (size_t)n0 * Ck, 32, 256, Ck);
  }

  f32x8 acc[4][4] = {};
  const int nsteps = Ck / 32;
  for (int s = 0; s < nsteps; ++s) {
    const int buf = s & 1;
    __syncthreads();                              // previous compute done; buf^1 free
    if (wv == 0) {
      if (s + 1 < nsteps) {
        const int kc = (s + 1) * 32;
        tdm_load_2d(ldsA[buf ^ 1], xb + (size_t)m0 * Ck + kc, 32, 128, Ck);
        tdm_load_2d(ldsB[buf ^ 1], wT + (size_t)n0 * Ck + kc, 32, 256, Ck);
        __builtin_amdgcn_s_wait_tensorcnt((short)2);   // current pair done (in-order)
      } else {
        __builtin_amdgcn_s_wait_tensorcnt((short)0);
      }
    }
    __syncthreads();                              // publish current buffers
    bf16x16 af[4], bfr[4];
#pragma unroll
    for (int mi = 0; mi < 4; ++mi) af[mi]  = lds_a_frag(ldsA[buf], wm + mi * 16, lane);
#pragma unroll
    for (int ni = 0; ni < 4; ++ni) bfr[ni] = lds_b_frag(ldsB[buf], wn + ni * 16, lane);
#pragma unroll
    for (int mi = 0; mi < 4; ++mi)
#pragma unroll
      for (int ni = 0; ni < 4; ++ni)
        acc[mi][ni] = wmma_bf16(af[mi], bfr[ni], acc[mi][ni]);
  }

  const int hl = lane >> 4, qn = lane & 15;
  const int m0w = m0 + wm, n0w = n0 + wn;
#pragma unroll
  for (int mi = 0; mi < 4; ++mi) {
#pragma unroll
    for (int ni = 0; ni < 4; ++ni) {
      int n = n0w + ni * 16 + qn;
      float bv = bias[n];
      int which = n >> 10;          // 0:q 1:k 2:v
      int cin = n & (Ck - 1);
      int hh = cin >> 6, dd = cin & 63;
#pragma unroll
      for (int e = 0; e < 8; ++e) {
        int tok = m0w + mi * 16 + hl * 8 + e;
        unsigned short v16 = f2bf(acc[mi][ni][e] + bv);
        int bb = tok >> 11, tt = tok & (Tk - 1);
        int bhh = bb * Hk + hh;
        if (which == 0)      qb[(bhh * Tk + tt) * Dk + dd] = v16;
        else if (which == 1) kb[(bhh * Tk + tt) * Dk + dd] = v16;
        else                 vT[(bhh * Dk + dd) * Tk + tt] = v16;
      }
    }
  }
}

// ---------------------------------------------------------------------------
// Transposed flash attention. One wave = one (b,h, 16-query tile).
// S^T = K * Q^T  (A = K rows=keys, B = Q^T -> k contiguous in q's D dim)
// O^T = V^T * P^T (A = vT rows=d, B = P^T via 16 ds_bpermute shuffles)
// Softmax stats are per-column (=query=lane%16): per-lane reduce + shfl_xor(16).
// ---------------------------------------------------------------------------
__global__ __launch_bounds__(256) void attn_kernel(
    const unsigned short* __restrict__ qbuf, const unsigned short* __restrict__ kbuf,
    const unsigned short* __restrict__ vTb, unsigned short* __restrict__ yb) {
  const int lane = threadIdx.x & 31;
  const int wid  = blockIdx.x * (blockDim.x >> 5) + (threadIdx.x >> 5);
  const int qt = wid & (Tk / 16 - 1);
  const int bh = wid >> 7;
  if (bh >= Bk * Hk) return;
  const int b = bh >> 4, hh = bh & (Hk - 1);
  const unsigned short* Q = qbuf + bh * Tk * Dk;
  const unsigned short* K = kbuf + bh * Tk * Dk;
  const unsigned short* V = vTb + bh * Dk * Tk;

  const int q0 = qt * 16;
  const int hl = lane >> 4, qn = lane & 15;
  const int qg = q0 + qn;

  bf16x16 Qb0 = load_b_frag(Q, Dk, q0, 0, lane);
  bf16x16 Qb1 = load_b_frag(Q, Dk, q0, 32, lane);

  f32x8 o[4] = {};
  float mrun = -1e30f, lrun = 0.f;
  const float sc = 0.125f * 1.44269504088896f;   // D^-1/2 * log2(e)

  const int nkb = (q0 >> 5) + 1;                 // 32-key blocks, causal bound
  for (int kbk = 0; kbk < nkb; ++kbk) {
    const int k0 = kbk * 32;
    if (kbk + 1 < nkb) {                         // prefetch next K/V tiles
      __builtin_prefetch(K + (k0 + 32) * Dk + lane * 8, 0, 1);
      __builtin_prefetch(V + (lane & 15) * Tk + k0 + 32, 0, 1);
    }
    bf16x16 a00 = load_a_frag(K, Dk, k0,      0,  lane);
    bf16x16 a01 = load_a_frag(K, Dk, k0,      32, lane);
    bf16x16 a10 = load_a_frag(K, Dk, k0 + 16, 0,  lane);
    bf16x16 a11 = load_a_frag(K, Dk, k0 + 16, 32, lane);
    f32x8 s0 = {}, s1 = {};
    s0 = wmma_bf16(a00, Qb0, s0); s0 = wmma_bf16(a01, Qb1, s0);
    s1 = wmma_bf16(a10, Qb0, s1); s1 = wmma_bf16(a11, Qb1, s1);

    float p0[8], p1[8];
    float tmax = -1e30f;
#pragma unroll
    for (int e = 0; e < 8; ++e) {
      int key0 = k0 + hl * 8 + e;
      int key1 = key0 + 16;
      p0[e] = (key0 <= qg) ? s0[e] * sc : -1e30f;
      p1[e] = (key1 <= qg) ? s1[e] * sc : -1e30f;
      tmax = fmaxf(tmax, fmaxf(p0[e], p1[e]));
    }
    tmax = fmaxf(tmax, __shfl_xor(tmax, 16, 32));
    float mnew = fmaxf(mrun, tmax);
    float rescale = exp2f(mrun - mnew);
    float rs = 0.f;
#pragma unroll
    for (int e = 0; e < 8; ++e) {
      p0[e] = exp2f(p0[e] - mnew);
      p1[e] = exp2f(p1[e] - mnew);
      rs += p0[e] + p1[e];
    }
    rs += __shfl_xor(rs, 16, 32);
    lrun = lrun * rescale + rs;
    mrun = mnew;
#pragma unroll
    for (int f4 = 0; f4 < 4; ++f4) o[f4] = o[f4] * rescale;

    // P^T C-layout -> B-fragment layout: elem i needs frag(i>>3)[i&7] from lane qn+16*hl
    bf16x16 Pb;
#pragma unroll
    for (int e = 0; e < 8; ++e) {
      float a0 = __shfl(p0[e], qn + 16 * hl, 32);
      float a1 = __shfl(p1[e], qn + 16 * hl, 32);
      Pb[e]     = f2bfh(a0);
      Pb[e + 8] = f2bfh(a1);
    }

#pragma unroll
    for (int f4 = 0; f4 < 4; ++f4) {
      bf16x16 va = load_a_frag(V, Tk, f4 * 16, k0, lane);
      o[f4] = wmma_bf16(va, Pb, o[f4]);
    }
  }

  const float inv = 1.0f / lrun;
  const int t = q0 + qn;
#pragma unroll
  for (int f4 = 0; f4 < 4; ++f4) {
    bf16x8 pk;
#pragma unroll
    for (int e = 0; e < 8; ++e) pk[e] = f2bfh(o[f4][e] * inv);
    int col = hh * Dk + f4 * 16 + hl * 8;
    *(bf16x8*)(yb + ((b * Tk + t) * Ck + col)) = pk;
  }
}

// ---------------------------------------------------------------------------
// Output projection: y[4096,1024] x Wproj[1024,1024] + bias -> fp32 out.
// Same TDM-staged block structure as qkv_gemm.
// ---------------------------------------------------------------------------
__global__ __launch_bounds__(256) void proj_gemm_kernel(
    const unsigned short* __restrict__ yb, const unsigned short* __restrict__ wT,
    const float* __restrict__ bias, float* __restrict__ out) {
  __shared__ unsigned short ldsA[2][128 * 32];
  __shared__ unsigned short ldsB[2][256 * 32];
  const int lane = threadIdx.x & 31;
  const int wv   = threadIdx.x >> 5;
  const int NTB = Ck / 256;                       // 4
  const int bm = blockIdx.x / NTB, bn = blockIdx.x % NTB;
  const int m0 = bm * 128, n0 = bn * 256;
  const int wm = (wv >> 2) * 64, wn = (wv & 3) * 64;

  if (wv == 0) {
    tdm_load_2d(ldsA[0], yb + (size_t)m0 * Ck, 32, 128, Ck);
    tdm_load_2d(ldsB[0], wT + (size_t)n0 * Ck, 32, 256, Ck);
  }

  f32x8 acc[4][4] = {};
  const int nsteps = Ck / 32;
  for (int s = 0; s < nsteps; ++s) {
    const int buf = s & 1;
    __syncthreads();
    if (wv == 0) {
      if (s + 1 < nsteps) {
        const int kc = (s + 1) * 32;
        tdm_load_2d(ldsA[buf ^ 1], yb + (size_t)m0 * Ck + kc, 32, 128, Ck);
        tdm_load_2d(ldsB[buf ^ 1], wT + (size_t)n0 * Ck + kc, 32, 256, Ck);
        __builtin_amdgcn_s_wait_tensorcnt((short)2);
      } else {
        __builtin_amdgcn_s_wait_tensorcnt((short)0);
      }
    }
    __syncthreads();
    bf16x16 af[4], bfr[4];
#pragma unroll
    for (int mi = 0; mi < 4; ++mi) af[mi]  = lds_a_frag(ldsA[buf], wm + mi * 16, lane);
#pragma unroll
    for (int ni = 0; ni < 4; ++ni) bfr[ni] = lds_b_frag(ldsB[buf], wn + ni * 16, lane);
#pragma unroll
    for (int mi = 0; mi < 4; ++mi)
#pragma unroll
      for (int ni = 0; ni < 4; ++ni)
        acc[mi][ni] = wmma_bf16(af[mi], bfr[ni], acc[mi][ni]);
  }

  const int hl = lane >> 4, qn = lane & 15;
  const int m0w = m0 + wm, n0w = n0 + wn;
#pragma unroll
  for (int mi = 0; mi < 4; ++mi) {
#pragma unroll
    for (int ni = 0; ni < 4; ++ni) {
      int n = n0w + ni * 16 + qn;
      float bv = bias[n];
#pragma unroll
      for (int e = 0; e < 8; ++e) {
        int tok = m0w + mi * 16 + hl * 8 + e;
        out[tok * Ck + n] = acc[mi][ni][e] + bv;
      }
    }
  }
}

// ---------------------------------------------------------------------------
extern "C" void kernel_launch(void* const* d_in, const int* in_sizes, int n_in,
                              void* d_out, int out_size, void* d_ws, size_t ws_size,
                              hipStream_t stream) {
  (void)in_sizes; (void)n_in; (void)out_size; (void)ws_size;
  const float* x     = (const float*)d_in[0];
  const float* Wqkv  = (const float*)d_in[1];
  const float* bqkv  = (const float*)d_in[2];
  const float* Wproj = (const float*)d_in[3];
  const float* bproj = (const float*)d_in[4];
  float* out = (float*)d_out;

  const int nX  = Bk * Tk * Ck;        // 4,194,304
  const int nWq = Ck * 3 * Ck;         // 3,145,728
  const int nWp = Ck * Ck;             // 1,048,576

  unsigned short* xb     = (unsigned short*)d_ws;
  unsigned short* wqkvT  = xb + nX;
  unsigned short* wprojT = wqkvT + nWq;
  unsigned short* qb     = wprojT + nWp;
  unsigned short* kb     = qb + nX;
  unsigned short* vT     = kb + nX;
  unsigned short* yb     = vT + nX;

  cvt_bf16_kernel<<<nX / 256, 256, 0, stream>>>(x, xb, nX);
  cvt_transpose_kernel<<<nWq / 256, 256, 0, stream>>>(Wqkv, wqkvT, Ck, 3 * Ck);
  cvt_transpose_kernel<<<nWp / 256, 256, 0, stream>>>(Wproj, wprojT, Ck, Ck);

  // 32 M-blocks x 12 N-blocks, 8 waves each (128x256 macro tiles)
  qkv_gemm_kernel<<<384, 256, 0, stream>>>(xb, wqkvT, bqkv, qb, kb, vT);

  // 32 (b,h) x 128 q-tiles = 4096 waves
  attn_kernel<<<512, 256, 0, stream>>>(qb, kb, vT, yb);

  // 32 M-blocks x 4 N-blocks
  proj_gemm_kernel<<<128, 256, 0, stream>>>(yb, wprojT, bproj, out);
}
